// NeRF_MLP_Compose_43774306681420
// MI455X (gfx1250) — compile-verified, matrix operand order checked by reading, not run on hardware
//
#include <hip/hip_runtime.h>
#include <hip/hip_bf16.h>

typedef __attribute__((ext_vector_type(8)))  _Float16 v8h;
typedef __attribute__((ext_vector_type(16))) _Float16 v16h;
typedef __attribute__((ext_vector_type(8)))  float    v8f;

#define NPTS      65536
#define NEXP      4
#define HID       256
#define KPE       96      // 84 padded to 96 (3 k-steps of 32)
#define DOUT      64
#define BM        128     // rows per tile
#define STRH      264     // LDS row stride in halves (256+8 pad -> conflict-free)
#define MAXT      516     // max tiles: 512 + 4 (padding)
#define NCAP      (NPTS + NEXP*BM)

__device__ __forceinline__ v8f wmma16(v16h a, v16h b, v8f c) {
  return __builtin_amdgcn_wmma_f32_16x16x32_f16(false, a, false, b, (short)0, c, false, false);
}

// Load a 16x32 f16 fragment from LDS (row-major, stride STRH halves).
// Works for both A (rows = M) and B (rows = N, i.e. W stored transposed).
__device__ __forceinline__ v16h load_frag(const _Float16* lds, int row0, int kbase, int lane) {
  int r  = lane & 15;
  int hi = (lane >> 4) & 1;
  const _Float16* p = lds + (row0 + r) * STRH + kbase + hi * 8;
  v8h lo = *(const v8h*)p;         // K = kbase + hi*8 .. +7
  v8h hh = *(const v8h*)(p + 16);  // K = kbase + 16 + hi*8 .. +7
  return __builtin_shufflevector(lo, hh, 0,1,2,3,4,5,6,7,8,9,10,11,12,13,14,15);
}

template<int ROWS, int KH>
__device__ __forceinline__ void stage_tile(_Float16* dst, const _Float16* __restrict__ src, int tid) {
  constexpr int CPR   = KH / 8;
  constexpr int TOTAL = ROWS * CPR;
  #pragma unroll 4
  for (int c = tid; c < TOTAL; c += 256) {
    int row = c / CPR;
    int cc  = c - row * CPR;
    *(v8h*)(dst + row * STRH + cc * 8) = *(const v8h*)(src + (size_t)row * KH + cc * 8);
  }
}

// ---------------- setup kernels ----------------

__global__ void init_kernel(int* counts) {           // zeroes counts[4] + cursors[4]
  int i = threadIdx.x;
  if (i < 8) counts[i] = 0;
}

__global__ void count_kernel(const int* __restrict__ lid, int* counts) {
  int i = blockIdx.x * 256 + threadIdx.x;
  if (i < NPTS) atomicAdd(&counts[lid[i]], 1);
}

__global__ void plan_kernel(const int* __restrict__ counts, int* off, int* nTiles,
                            int* tileE, int* tileB, int* perm) {
  if (threadIdx.x != 0 || blockIdx.x != 0) return;
  int o = 0, t = 0;
  for (int e = 0; e < NEXP; ++e) {
    off[e] = o;
    int c  = counts[e];
    int nt = (c + BM - 1) >> 7;
    for (int i = 0; i < nt; ++i) { tileE[t] = e; tileB[t] = o + i * BM; ++t; }
    int pc = nt << 7;
    for (int p = c; p < pc; ++p) perm[o + p] = -1;   // padded rows -> discard
    o += pc;
  }
  *nTiles = t;
}

__global__ void pe_scatter(const float* __restrict__ x, const int* __restrict__ lid,
                           const int* __restrict__ off, int* cursors,
                           int* perm, _Float16* __restrict__ xe) {
  int i = blockIdx.x * 256 + threadIdx.x;
  if (i >= NPTS) return;
  int e   = lid[i];
  int pos = off[e] + atomicAdd(&cursors[e], 1);
  perm[pos] = i;
  float x3 = x[i * 4 + 3];
  float xv[4] = { x[i * 4] / x3, x[i * 4 + 1] / x3, x[i * 4 + 2] / x3, x3 };
  _Float16* row = xe + (size_t)pos * KPE;
  row[0] = (_Float16)xv[0]; row[1] = (_Float16)xv[1];
  row[2] = (_Float16)xv[2]; row[3] = (_Float16)xv[3];
  int idx = 4;
  float f = 3.14159265358979323846f;   // 2^0 * pi
  for (int fi = 0; fi < 10; ++fi) {
    #pragma unroll
    for (int j = 0; j < 4; ++j) {
      float a = xv[j] * f;
      row[idx++] = (_Float16)sinf(a);
      row[idx++] = (_Float16)cosf(a);
    }
    f *= 2.0f;
  }
  for (; idx < KPE; ++idx) row[idx] = (_Float16)0.0f;
}

// ---------------- weight convert/transpose to f16 [n][k] ----------------

__global__ void conv_w0(const float* __restrict__ W0, _Float16* __restrict__ wt0) {
  int idx = blockIdx.x * 256 + threadIdx.x;           // 4*256*96
  if (idx >= NEXP * HID * KPE) return;
  int e   = idx / (HID * KPE);
  int rem = idx - e * HID * KPE;
  int n   = rem / KPE;
  int kk  = rem - n * KPE;
  wt0[idx] = (kk < 84) ? (_Float16)W0[((size_t)e * 84 + kk) * HID + n] : (_Float16)0.0f;
}

__global__ void conv_wh(const float* __restrict__ Wh, _Float16* __restrict__ wth) {
  int idx = blockIdx.x * 256 + threadIdx.x;           // 12*256*256
  if (idx >= NEXP * 3 * HID * HID) return;
  int el  = idx / (HID * HID);
  int rem = idx - el * HID * HID;
  int n   = rem / HID;
  int k   = rem - n * HID;
  wth[idx] = (_Float16)Wh[((size_t)el * HID + k) * HID + n];
}

__global__ void conv_wout(const float* __restrict__ Wout, _Float16* __restrict__ wto) {
  int idx = blockIdx.x * 256 + threadIdx.x;           // 4*64*256
  if (idx >= NEXP * DOUT * HID) return;
  int e   = idx / (DOUT * HID);
  int rem = idx - e * DOUT * HID;
  int n   = rem / HID;
  int k   = rem - n * HID;
  wto[idx] = (_Float16)Wout[((size_t)e * HID + k) * DOUT + n];
}

// ---------------- main MLP kernel: 1 tile (128 rows, one expert) per block ----------------

__global__ __launch_bounds__(256)
void mlp_kernel(const _Float16* __restrict__ xe, const int* __restrict__ perm,
                const int* __restrict__ tileE, const int* __restrict__ tileB,
                const int* __restrict__ nTiles,
                const _Float16* __restrict__ wt0, const _Float16* __restrict__ wth,
                const _Float16* __restrict__ wto,
                const float* __restrict__ b0, const float* __restrict__ bh,
                const float* __restrict__ scal, const float* __restrict__ bout,
                const float* __restrict__ in_dim, float* __restrict__ out) {
  __shared__ _Float16 sh_h[BM * STRH];    // activations (in-place per wave)
  __shared__ _Float16 sh_w[BM * STRH];    // weight chunk [128][K]

  int t = blockIdx.x;
  if (t >= *nTiles) return;
  int e    = tileE[t];
  int base = tileB[t];

  int tid  = threadIdx.x;
  int wv   = tid >> 5;
  int lane = tid & 31;
  int ncol = lane & 15;
  int hi   = (lane >> 4) & 1;
  int m0   = wv * 16;

  // stage xe tile into activation buffer (cols 0..95)
  stage_tile<BM, KPE>(sh_h, xe + (size_t)base * KPE, tid);

  v8f zero = (v8f)0.0f;
  v8f acc[16];

  // ---- layer 0: h = relu(xe @ W0 + b0), K = 96 ----
  #pragma unroll
  for (int j = 0; j < 16; ++j) acc[j] = zero;
  {
    const _Float16* w0e = wt0 + (size_t)e * HID * KPE;
    #pragma unroll
    for (int chunk = 0; chunk < 2; ++chunk) {
      __syncthreads();
      stage_tile<BM, KPE>(sh_w, w0e + (size_t)chunk * BM * KPE, tid);
      __syncthreads();
      #pragma unroll
      for (int ks = 0; ks < 3; ++ks) {
        v16h a = load_frag(sh_h, m0, ks * 32, lane);
        #pragma unroll
        for (int j = 0; j < 8; ++j) {
          v16h b = load_frag(sh_w, j * 16, ks * 32, lane);
          acc[chunk * 8 + j] = wmma16(a, b, acc[chunk * 8 + j]);
        }
      }
    }
    const float* b0e = b0 + e * HID;
    #pragma unroll
    for (int j = 0; j < 16; ++j) {
      int n = j * 16 + ncol;
      float bias = b0e[n];
      #pragma unroll
      for (int r = 0; r < 8; ++r) {
        float v = acc[j][r] + bias;
        v = v > 0.0f ? v : 0.0f;
        sh_h[(m0 + r + 8 * hi) * STRH + n] = (_Float16)v;
      }
    }
  }

  // ---- 3 residual blocks: h = s*relu(h @ Wh + bh) + h, K = 256 ----
  for (int L = 0; L < 3; ++L) {
    #pragma unroll
    for (int j = 0; j < 16; ++j) acc[j] = zero;
    const _Float16* wl = wth + (size_t)(e * 3 + L) * HID * HID;
    #pragma unroll
    for (int chunk = 0; chunk < 2; ++chunk) {
      __syncthreads();
      stage_tile<BM, HID>(sh_w, wl + (size_t)chunk * BM * HID, tid);
      __syncthreads();
      #pragma unroll
      for (int ks = 0; ks < 8; ++ks) {
        v16h a = load_frag(sh_h, m0, ks * 32, lane);
        #pragma unroll
        for (int j = 0; j < 8; ++j) {
          v16h b = load_frag(sh_w, j * 16, ks * 32, lane);
          acc[chunk * 8 + j] = wmma16(a, b, acc[chunk * 8 + j]);
        }
      }
    }
    float s = scal[e * 3 + L];
    const float* bl = bh + (e * 3 + L) * HID;
    #pragma unroll
    for (int j = 0; j < 16; ++j) {
      int n = j * 16 + ncol;
      float bias = bl[n];
      #pragma unroll
      for (int r = 0; r < 8; ++r) {
        int row = m0 + r + 8 * hi;
        float old = (float)sh_h[row * STRH + n];
        float v = acc[j][r] + bias;
        v = v > 0.0f ? v : 0.0f;
        sh_h[row * STRH + n] = (_Float16)(s * v + old);
      }
    }
  }

  // ---- output layer: out = (h @ Wout + bout) / in_dim, N = 64, K = 256 ----
  v8f occ[4];
  #pragma unroll
  for (int j = 0; j < 4; ++j) occ[j] = zero;
  __syncthreads();
  stage_tile<DOUT, HID>(sh_w, wto + (size_t)e * DOUT * HID, tid);
  __syncthreads();
  #pragma unroll
  for (int ks = 0; ks < 8; ++ks) {
    v16h a = load_frag(sh_h, m0, ks * 32, lane);
    #pragma unroll
    for (int j = 0; j < 4; ++j) {
      v16h b = load_frag(sh_w, j * 16, ks * 32, lane);
      occ[j] = wmma16(a, b, occ[j]);
    }
  }
  const float* boe = bout + e * DOUT;
  #pragma unroll
  for (int j = 0; j < 4; ++j) {
    int n = j * 16 + ncol;
    float bias = boe[n];
    #pragma unroll
    for (int r = 0; r < 8; ++r) {
      int row = m0 + r + 8 * hi;
      int g   = perm[base + row];
      if (g >= 0) {
        out[(size_t)g * DOUT + n] = (occ[j][r] + bias) / in_dim[g];
      }
    }
  }
}

// ---------------- host launcher ----------------

extern "C" void kernel_launch(void* const* d_in, const int* in_sizes, int n_in,
                              void* d_out, int out_size, void* d_ws, size_t ws_size,
                              hipStream_t stream) {
  const float* x      = (const float*)d_in[0];
  const float* in_dim = (const float*)d_in[1];
  const int*   lid    = (const int*)d_in[2];
  const float* W0     = (const float*)d_in[3];
  const float* b0     = (const float*)d_in[4];
  const float* Wh     = (const float*)d_in[5];
  const float* bh     = (const float*)d_in[6];
  const float* scal   = (const float*)d_in[7];
  const float* Wout   = (const float*)d_in[8];
  const float* bout   = (const float*)d_in[9];
  float* out          = (float*)d_out;

  char* ws     = (char*)d_ws;
  int* counts  = (int*)ws;         // 4
  int* cursors = counts + 4;       // 4
  int* off     = cursors + 4;      // 4
  int* nTiles  = off + 4;          // 1 (+3 pad)
  int* tileE   = nTiles + 4;       // MAXT
  int* tileB   = tileE + MAXT;     // MAXT
  int* perm    = tileB + MAXT;     // NCAP

  size_t o = (size_t)((char*)(perm + NCAP) - ws);
  o = (o + 255) & ~(size_t)255;
  _Float16* xe  = (_Float16*)(ws + o); o += (size_t)NCAP * KPE * 2;
  _Float16* wt0 = (_Float16*)(ws + o); o += (size_t)NEXP * HID * KPE * 2;
  _Float16* wth = (_Float16*)(ws + o); o += (size_t)NEXP * 3 * HID * HID * 2;
  _Float16* wto = (_Float16*)(ws + o); o += (size_t)NEXP * DOUT * HID * 2;

  init_kernel<<<1, 32, 0, stream>>>(counts);
  count_kernel<<<NPTS / 256, 256, 0, stream>>>(lid, counts);
  plan_kernel<<<1, 1, 0, stream>>>(counts, off, nTiles, tileE, tileB, perm);
  conv_w0<<<(NEXP * HID * KPE + 255) / 256, 256, 0, stream>>>(W0, wt0);
  conv_wh<<<(NEXP * 3 * HID * HID + 255) / 256, 256, 0, stream>>>(Wh, wth);
  conv_wout<<<(NEXP * DOUT * HID + 255) / 256, 256, 0, stream>>>(Wout, wto);
  pe_scatter<<<NPTS / 256, 256, 0, stream>>>(x, lid, off, cursors, perm, xe);
  mlp_kernel<<<MAXT, 256, 0, stream>>>(xe, perm, tileE, tileB, nTiles,
                                       wt0, wth, wto, b0, bh, scal, bout, in_dim, out);
}
